// JSCCDecoder_59657095741751
// MI455X (gfx1250) — matrix operand order repaired and checked by reading, not compiled, not run on hardware
//
#include <hip/hip_runtime.h>

// ---------------------------------------------------------------------------
// JSCC Swin decoder for MI455X (gfx1250, wave32, WMMA).
// All GEMM-class math on V_WMMA_F32_16X16X32_F16 (f16 in, f32 acc).
// B operands pre-swizzled into WMMA lane order -> every tile load is b128.
// Attention q/k staged to LDS via async global->LDS copies (ASYNCcnt).
// ---------------------------------------------------------------------------

typedef __attribute__((ext_vector_type(16))) _Float16 v16h;
typedef __attribute__((ext_vector_type(8)))  _Float16 v8h;
typedef __attribute__((ext_vector_type(8)))  float    v8f;
typedef int v4i __attribute__((vector_size(16)));

#define T_TOK   32768      // B*H*W tokens
#define C_DIM   256
#define L_DIM   4096       // H*W
#define H_DIM   64
#define NHEADS  8

#if defined(__HIP_DEVICE_COMPILE__) &&                                        \
    __has_builtin(__builtin_amdgcn_global_load_async_to_lds_b128) &&          \
    __has_builtin(__builtin_amdgcn_s_wait_asynccnt)
#define HAVE_ASYNC 1
#else
#define HAVE_ASYNC 0
#endif

#if HAVE_ASYNC
// 16-byte global -> LDS async copy (tracked by ASYNCcnt).
// Signature (from clang diagnostic): (v4i addrspace(1)*, v4i addrspace(3)*,
//                                     imm offset, imm cpol)
#define ASYNC_CP16(ldsp, gp)                                                  \
  __builtin_amdgcn_global_load_async_to_lds_b128(                             \
      (__attribute__((address_space(1))) v4i*)(gp),                           \
      (__attribute__((address_space(3))) v4i*)(ldsp), 0, 0)
#endif

__device__ inline v8f wmma_ff16(v16h a, v16h b, v8f c) {
  return __builtin_amdgcn_wmma_f32_16x16x32_f16(false, a, false, b, (short)0, c,
                                                false, false);
}

__device__ inline v16h cat8(v8h lo, v8h hi) {
  return __builtin_shufflevector(lo, hi, 0, 1, 2, 3, 4, 5, 6, 7, 8, 9, 10, 11,
                                 12, 13, 14, 15);
}

// A tile 16x32 (MxK), row-major source (16B-aligned rows). Contiguous k order (B packed the same):
// lanes 0-15: row=lane, K={0..15}; lanes 16-31: row=lane-16, K={16..31}
__device__ inline v16h load_a_tile(const _Float16* A, int lda) {
  const int lane = threadIdx.x & 31;
  const _Float16* p = A + (size_t)(lane & 15) * lda + (lane >> 4) * 16;
  return cat8(*(const v8h*)p, *(const v8h*)(p + 8));
}

// B tile from packed storage: tile = 512 f16, lane's 16 values contiguous.
__device__ inline v16h load_b_packed(const _Float16* tile) {
  const _Float16* p = tile + (threadIdx.x & 31) * 16;
  return cat8(*(const v8h*)p, *(const v8h*)(p + 8));
}

// ---------------------------------------------------------------------------
// Conversion / packing kernels
// ---------------------------------------------------------------------------

// x: (B=8, R=256, L=4096) fp32 -> xh: (B, L, R) f16
__global__ __launch_bounds__(256) void cvt_x_kernel(const float* __restrict__ x,
                                                    _Float16* __restrict__ xo) {
  size_t i = (size_t)blockIdx.x * blockDim.x + threadIdx.x;
  if (i >= (size_t)8 * 256 * 4096) return;
  int l = (int)(i % 4096);
  size_t q = i / 4096;
  int r = (int)(q % 256);
  int b = (int)(q / 256);
  xo[((size_t)b * 4096 + l) * 256 + r] = (_Float16)x[i];
}

// Pack W (Nout, K) row-major fp32 into B-tile-packed f16 for B = W^T (K x Nout).
// Packed: tile (tn, tk) at ((tn*(K/32)+tk)*512), element lane*16+j holds
// B[tk*32 + (lane>>4)*16 + j][tn*16 + (lane&15)] = W[n*K + k].
__global__ __launch_bounds__(256) void cvt_pack_wT_kernel(
    const float* __restrict__ w, _Float16* __restrict__ o, int Nout, int K) {
  size_t per = (size_t)Nout * K;
  const float* src = w + (size_t)blockIdx.z * per;
  _Float16* dst = o + (size_t)blockIdx.z * per;
  size_t i = (size_t)blockIdx.x * blockDim.x + threadIdx.x;
  if (i >= per) return;
  int j = (int)(i & 15);
  int lane = (int)((i >> 4) & 31);
  int tile = (int)(i >> 9);
  int ktiles = K >> 5;
  int tk = tile % ktiles, tn = tile / ktiles;
  int n = tn * 16 + (lane & 15);
  int k = tk * 32 + (lane >> 4) * 16 + j;
  dst[i] = (_Float16)src[(size_t)n * K + k];
}

// Pack W already stored (K, N) row-major (ra_weight expert) into B-tile-packed.
__global__ __launch_bounds__(256) void cvt_pack_kn_kernel(
    const float* __restrict__ w, _Float16* __restrict__ o, int K, int N) {
  size_t per = (size_t)K * N;
  const float* src = w + (size_t)blockIdx.z * per;
  _Float16* dst = o + (size_t)blockIdx.z * per;
  size_t i = (size_t)blockIdx.x * blockDim.x + threadIdx.x;
  if (i >= per) return;
  int j = (int)(i & 15);
  int lane = (int)((i >> 4) & 31);
  int tile = (int)(i >> 9);
  int ktiles = K >> 5;
  int tk = tile % ktiles, tn = tile / ktiles;
  int n = tn * 16 + (lane & 15);
  int k = tk * 32 + (lane >> 4) * 16 + j;
  dst[i] = (_Float16)src[(size_t)k * N + n];
}

// ---------------------------------------------------------------------------
// Rate adaption: dense 3-expert GEMM, predicated epilogue on per-token index.
// ---------------------------------------------------------------------------
__global__ __launch_bounds__(256) void ra_gemm_kernel(
    const _Float16* __restrict__ A, const _Float16* __restrict__ Wp,
    const float* __restrict__ bias, const float* __restrict__ rtok,
    const int* __restrict__ idx, float* y) {
  const int e = blockIdx.z;
  const _Float16* Bp = Wp + (size_t)e * 256 * 256;
  const int wave = threadIdx.x >> 5, lane = threadIdx.x & 31;
  const int nn = lane & 15, hf = lane >> 4;
  const int m0 = (blockIdx.y * 8 + wave) * 16;
  const int n0 = blockIdx.x * 64;
  v8f acc[4] = {};
  for (int k0 = 0; k0 < 256; k0 += 32) {
    v16h a = load_a_tile(A + (size_t)m0 * 256 + k0, 256);
#pragma unroll
    for (int t = 0; t < 4; ++t) {
      const _Float16* bt =
          Bp + (((size_t)(n0 >> 4) + t) * (256 >> 5) + (k0 >> 5)) * 512;
      acc[t] = wmma_ff16(a, load_b_packed(bt), acc[t]);
    }
  }
#pragma unroll
  for (int t = 0; t < 4; ++t)
#pragma unroll
    for (int j = 0; j < 8; ++j) {
      int row = m0 + j + hf * 8;
      int col = n0 + t * 16 + nn;
      if (idx[row] == e)
        y[(size_t)row * 256 + col] =
            acc[t][j] + bias[e * 256 + col] + rtok[e * 256 + col];
    }
}

// ---------------------------------------------------------------------------
// LayerNorm over C=256, one wave per token, f16 output.
// ---------------------------------------------------------------------------
__global__ __launch_bounds__(256) void ln_kernel(const float* __restrict__ y,
                                                 const float* __restrict__ g,
                                                 const float* __restrict__ be,
                                                 _Float16* __restrict__ out) {
  const int wave = threadIdx.x >> 5, lane = threadIdx.x & 31;
  const int t = blockIdx.x * 8 + wave;
  const float* p = y + (size_t)t * 256;
  float v[8], s = 0.f, sq = 0.f;
#pragma unroll
  for (int i = 0; i < 8; ++i) {
    v[i] = p[lane * 8 + i];
    s += v[i];
    sq += v[i] * v[i];
  }
  for (int off = 1; off < 32; off <<= 1) {
    s  += __shfl_xor(s, off, 32);
    sq += __shfl_xor(sq, off, 32);
  }
  float mean = s * (1.f / 256.f);
  float var  = sq * (1.f / 256.f) - mean * mean;
  float rs   = rsqrtf(var + 1e-5f);
#pragma unroll
  for (int i = 0; i < 8; ++i) {
    int c = lane * 8 + i;
    out[(size_t)t * 256 + c] = (_Float16)((v[i] - mean) * rs * g[c] + be[c]);
  }
}

// ---------------------------------------------------------------------------
// Generic f16 GEMM: A(MxK) row-major, B tile-packed, fp32 acc.
// EPI: 0 = bias -> f16; 1 = bias + residual -> fp32; 2 = bias + GELU -> f16.
// Wave tile 16x64, 8 waves / block.
// ---------------------------------------------------------------------------
template <int EPI>
__global__ __launch_bounds__(256) void gemm16(const _Float16* A,
                                              const _Float16* Bp,
                                              const float* bias,
                                              const float* resid, void* outp,
                                              int M, int N, int K) {
  const int wave = threadIdx.x >> 5, lane = threadIdx.x & 31;
  const int nn = lane & 15, hf = lane >> 4;
  const int m0 = (blockIdx.y * 8 + wave) * 16;
  const int n0 = blockIdx.x * 64;
  if (m0 >= M) return;
  const int ktiles = K >> 5;
  v8f acc[4] = {};
  for (int k0 = 0; k0 < K; k0 += 32) {
    __builtin_prefetch(A + (size_t)m0 * K + k0 + 256, 0, 0);  // global_prefetch
    v16h a = load_a_tile(A + (size_t)m0 * K + k0, K);
#pragma unroll
    for (int t = 0; t < 4; ++t) {
      const _Float16* bt =
          Bp + (((size_t)(n0 >> 4) + t) * ktiles + (k0 >> 5)) * 512;
      acc[t] = wmma_ff16(a, load_b_packed(bt), acc[t]);
    }
  }
#pragma unroll
  for (int t = 0; t < 4; ++t)
#pragma unroll
    for (int j = 0; j < 8; ++j) {
      int row = m0 + j + hf * 8;
      int col = n0 + t * 16 + nn;
      float v = acc[t][j] + bias[col];
      if constexpr (EPI == 0) {
        ((_Float16*)outp)[(size_t)row * N + col] = (_Float16)v;
      } else if constexpr (EPI == 1) {
        ((float*)outp)[(size_t)row * N + col] =
            resid[(size_t)row * N + col] + v;
      } else {
        float gv = 0.5f * v * (1.0f + erff(v * 0.70710678118654752f));
        ((_Float16*)outp)[(size_t)row * N + col] = (_Float16)gv;
      }
    }
}

// ---------------------------------------------------------------------------
// Windowed attention, one workgroup per (window, head). dh=32 == WMMA K.
// q staged row-major (A operand); k and v staged pre-swizzled into the packed
// B-tile layout, so every WMMA operand read from LDS is a pair of b128 loads.
// q/k staging uses async global->LDS copies (ASYNCcnt) when available.
// ---------------------------------------------------------------------------
template <int WS, int NWAVES>
__global__ __launch_bounds__(NWAVES * 32) void attn_kernel(
    const _Float16* __restrict__ qkv, const float* __restrict__ rpb,
    _Float16* __restrict__ o) {
  constexpr int N = WS * WS;
  constexpr int NT = N / 16;
  constexpr int SLABS = NT / NWAVES;
  constexpr int NW = H_DIM / WS;

  __shared__ _Float16 qLds[N * 32];            // [n][d] row-major
  __shared__ _Float16 kP[NT * 512];            // k^T B-tiles (scores)
  __shared__ _Float16 vP[(N / 32) * 2 * 512];  // V B-tiles (P @ V)
  __shared__ _Float16 pLds[NWAVES * 16 * N];   // per-wave P staging

  const int hd = blockIdx.y;
  const int g  = blockIdx.x;
  const int b  = g / (NW * NW);
  const int rr = g % (NW * NW);
  const int wi = rr / NW, wj = rr % NW;

  // ---- stage q/k/v for this (window, head): one thread per token ----
  if (threadIdx.x < N) {
    const int n = threadIdx.x;
    const int t = b * L_DIM + (wi * WS + n / WS) * H_DIM + wj * WS + (n % WS);
    const _Float16* src = qkv + (size_t)t * 768 + hd * 32;
    _Float16* qdst  = qLds + n * 32;
    _Float16* kdst0 = kP + (n >> 4) * 512 + (n & 15) * 16;        // d 0..15
    _Float16* kdst1 = kdst0 + 256;                                 // d 16..31
#if HAVE_ASYNC
    ASYNC_CP16(qdst,      src);
    ASYNC_CP16(qdst + 8,  src + 8);
    ASYNC_CP16(qdst + 16, src + 16);
    ASYNC_CP16(qdst + 24, src + 24);
    ASYNC_CP16(kdst0,     src + 256);
    ASYNC_CP16(kdst0 + 8, src + 264);
    ASYNC_CP16(kdst1,     src + 272);
    ASYNC_CP16(kdst1 + 8, src + 280);
#else
    *(v8h*)qdst        = *(const v8h*)(src);
    *(v8h*)(qdst + 8)  = *(const v8h*)(src + 8);
    *(v8h*)(qdst + 16) = *(const v8h*)(src + 16);
    *(v8h*)(qdst + 24) = *(const v8h*)(src + 24);
    *(v8h*)kdst0       = *(const v8h*)(src + 256);
    *(v8h*)(kdst0 + 8) = *(const v8h*)(src + 264);
    *(v8h*)kdst1       = *(const v8h*)(src + 272);
    *(v8h*)(kdst1 + 8) = *(const v8h*)(src + 280);
#endif
    // V: vectorized global read, register->LDS swizzle into B-tile layout.
    // value V[key=n][d] -> tile ((n>>5)*2 + (d>>4)), lane=((n>>4)&1)*16+(d&15), j=n&15
#pragma unroll
    for (int c = 0; c < 4; ++c) {
      v8h vv = *(const v8h*)(src + 512 + c * 8);
#pragma unroll
      for (int e = 0; e < 8; ++e) {
        int d = c * 8 + e;
        vP[((n >> 5) * 2 + (d >> 4)) * 512 +
           ((((n >> 4) & 1) * 16 + (d & 15)) * 16) + (n & 15)] = vv[e];
      }
    }
  }
#if HAVE_ASYNC
  __builtin_amdgcn_s_wait_asynccnt(0);
#endif
  __syncthreads();

  const int wave = threadIdx.x >> 5, lane = threadIdx.x & 31;
  const int nn = lane & 15, hf = lane >> 4;
  _Float16* pw = pLds + (size_t)wave * 16 * N;
  const float scale = 0.17677669529663687f;  // dh^-0.5

  for (int s = 0; s < SLABS; ++s) {
    const int m0 = (wave * SLABS + s) * 16;

    // scores slab (16 x N) = q_slab @ k^T
    v8f acc[NT] = {};
    v16h a = load_a_tile(qLds + (size_t)m0 * 32, 32);
#pragma unroll
    for (int nt = 0; nt < NT; ++nt)
      acc[nt] = wmma_ff16(a, load_b_packed(kP + nt * 512), acc[nt]);

    // scale + relative position bias
#pragma unroll
    for (int nt = 0; nt < NT; ++nt)
#pragma unroll
      for (int j = 0; j < 8; ++j) {
        int qi = m0 + j + hf * 8;
        int ki = nt * 16 + nn;
        int ry = qi / WS - ki / WS + (WS - 1);
        int rx = qi % WS - ki % WS + (WS - 1);
        acc[nt][j] = acc[nt][j] * scale +
                     rpb[(ry * (2 * WS - 1) + rx) * NHEADS + hd];
      }

    // softmax: each 16-lane half holds complete rows (j and j+8); xor<16
    // shuffles never cross the half boundary
#pragma unroll
    for (int j = 0; j < 8; ++j) {
      float mx = -1e30f;
#pragma unroll
      for (int nt = 0; nt < NT; ++nt) mx = fmaxf(mx, acc[nt][j]);
      for (int off = 1; off < 16; off <<= 1)
        mx = fmaxf(mx, __shfl_xor(mx, off, 32));
      float sum = 0.f;
#pragma unroll
      for (int nt = 0; nt < NT; ++nt) {
        float p = __expf(acc[nt][j] - mx);
        acc[nt][j] = p;
        sum += p;
      }
      for (int off = 1; off < 16; off <<= 1) sum += __shfl_xor(sum, off, 32);
      float inv = 1.0f / sum;
#pragma unroll
      for (int nt = 0; nt < NT; ++nt)
        pw[(size_t)(j + hf * 8) * N + nt * 16 + nn] =
            (_Float16)(acc[nt][j] * inv);
    }

    // O slab (16 x 32) = P_slab @ V   (same-wave LDS ops are in-order)
    v8f oacc[2] = {};
    for (int k0 = 0; k0 < N; k0 += 32) {
      v16h pa = load_a_tile(pw + k0, N);
#pragma unroll
      for (int dt = 0; dt < 2; ++dt)
        oacc[dt] = wmma_ff16(
            pa, load_b_packed(vP + ((k0 >> 5) * 2 + dt) * 512), oacc[dt]);
    }
#pragma unroll
    for (int dt = 0; dt < 2; ++dt)
#pragma unroll
      for (int j = 0; j < 8; ++j) {
        int qi = m0 + j + hf * 8;
        int t = b * L_DIM + (wi * WS + qi / WS) * H_DIM + wj * WS + (qi % WS);
        o[(size_t)t * C_DIM + hd * 32 + dt * 16 + nn] = (_Float16)oacc[dt][j];
      }
  }
}

// ---------------------------------------------------------------------------
// Final (B, L, C) fp32 -> (B, C, H, W) fp32
// ---------------------------------------------------------------------------
__global__ __launch_bounds__(256) void out_kernel(const float* __restrict__ y,
                                                  float* __restrict__ out) {
  size_t i = (size_t)blockIdx.x * blockDim.x + threadIdx.x;
  if (i >= (size_t)8 * 256 * 4096) return;
  int l = (int)(i % 4096);
  size_t q = i / 4096;
  int c = (int)(q % 256);
  int b = (int)(q / 256);
  out[i] = y[((size_t)b * 4096 + l) * 256 + c];
}

// ---------------------------------------------------------------------------
// Host orchestration
// ---------------------------------------------------------------------------
extern "C" void kernel_launch(void* const* d_in, const int* in_sizes, int n_in,
                              void* d_out, int out_size, void* d_ws,
                              size_t ws_size, hipStream_t stream) {
  const float* x        = (const float*)d_in[0];
  const int*   indexes  = (const int*)d_in[1];
  const float* ra_w     = (const float*)d_in[2];
  const float* ra_b     = (const float*)d_in[3];
  const float* rate_tok = (const float*)d_in[4];
  const float* ln1_g    = (const float*)d_in[5];
  const float* ln1_b    = (const float*)d_in[6];
  const float* qkv_w    = (const float*)d_in[7];
  const float* qkv_b    = (const float*)d_in[8];
  const float* proj_w   = (const float*)d_in[9];
  const float* proj_b   = (const float*)d_in[10];
  const float* ln2_g    = (const float*)d_in[11];
  const float* ln2_b    = (const float*)d_in[12];
  const float* fc1_w    = (const float*)d_in[13];
  const float* fc1_b    = (const float*)d_in[14];
  const float* fc2_w    = (const float*)d_in[15];
  const float* fc2_b    = (const float*)d_in[16];
  const float* rpb[3]   = {(const float*)d_in[17], (const float*)d_in[18],
                           (const float*)d_in[19]};

  char* ws = (char*)d_ws;
  const size_t OFF_Y    = 0;
  const size_t OFF_XH   = OFF_Y    + (size_t)T_TOK * 256 * 4;
  const size_t OFF_LNH  = OFF_XH   + (size_t)T_TOK * 256 * 2;
  const size_t OFF_QKVH = OFF_LNH  + (size_t)T_TOK * 256 * 2;
  const size_t OFF_OH   = OFF_QKVH + (size_t)T_TOK * 768 * 2;
  const size_t OFF_HH   = OFF_OH   + (size_t)T_TOK * 256 * 2;
  const size_t OFF_RAW  = OFF_HH   + (size_t)T_TOK * 1024 * 2;
  const size_t OFF_QKVW = OFF_RAW  + (size_t)3 * 256 * 256 * 2;
  const size_t OFF_PRJW = OFF_QKVW + (size_t)3 * 256 * 768 * 2;
  const size_t OFF_FC1W = OFF_PRJW + (size_t)3 * 256 * 256 * 2;
  const size_t OFF_FC2W = OFF_FC1W + (size_t)3 * 256 * 1024 * 2;

  float*     y     = (float*)(ws + OFF_Y);
  _Float16*  xh    = (_Float16*)(ws + OFF_XH);
  _Float16*  lnh   = (_Float16*)(ws + OFF_LNH);
  _Float16*  qkvh  = (_Float16*)(ws + OFF_QKVH);
  _Float16*  oh    = (_Float16*)(ws + OFF_OH);
  _Float16*  hh    = (_Float16*)(ws + OFF_HH);
  _Float16*  rawp  = (_Float16*)(ws + OFF_RAW);
  _Float16*  qkvwp = (_Float16*)(ws + OFF_QKVW);
  _Float16*  prjwp = (_Float16*)(ws + OFF_PRJW);
  _Float16*  fc1wp = (_Float16*)(ws + OFF_FC1W);
  _Float16*  fc2wp = (_Float16*)(ws + OFF_FC2W);

  // ---- conversions + B-tile packing ----
  cvt_x_kernel<<<32768, 256, 0, stream>>>(x, xh);
  cvt_pack_kn_kernel<<<dim3(256, 1, 3),  256, 0, stream>>>(ra_w, rawp, 256, 256);
  cvt_pack_wT_kernel<<<dim3(768, 1, 3),  256, 0, stream>>>(qkv_w, qkvwp, 768, 256);
  cvt_pack_wT_kernel<<<dim3(256, 1, 3),  256, 0, stream>>>(proj_w, prjwp, 256, 256);
  cvt_pack_wT_kernel<<<dim3(1024, 1, 3), 256, 0, stream>>>(fc1_w, fc1wp, 1024, 256);
  cvt_pack_wT_kernel<<<dim3(1024, 1, 3), 256, 0, stream>>>(fc2_w, fc2wp, 256, 1024);

  // ---- rate adaption ----
  ra_gemm_kernel<<<dim3(4, 256, 3), 256, 0, stream>>>(xh, rawp, ra_b, rate_tok,
                                                      indexes, y);

  // ---- 3 Swin layers ----
  for (int i = 0; i < 3; ++i) {
    ln_kernel<<<4096, 256, 0, stream>>>(y, ln1_g + i * 256, ln1_b + i * 256, lnh);
    gemm16<0><<<dim3(12, 256), 256, 0, stream>>>(
        lnh, qkvwp + (size_t)i * 256 * 768, qkv_b + i * 768, nullptr, qkvh,
        T_TOK, 768, 256);
    if (i == 0)
      attn_kernel<8, 4><<<dim3(512, 8), 128, 0, stream>>>(qkvh, rpb[i], oh);
    else
      attn_kernel<16, 8><<<dim3(128, 8), 256, 0, stream>>>(qkvh, rpb[i], oh);
    gemm16<1><<<dim3(4, 256), 256, 0, stream>>>(
        oh, prjwp + (size_t)i * 256 * 256, proj_b + i * 256, y, y,
        T_TOK, 256, 256);
    ln_kernel<<<4096, 256, 0, stream>>>(y, ln2_g + i * 256, ln2_b + i * 256, lnh);
    gemm16<2><<<dim3(16, 256), 256, 0, stream>>>(
        lnh, fc1wp + (size_t)i * 256 * 1024, fc1_b + i * 1024, nullptr, hh,
        T_TOK, 1024, 256);
    gemm16<1><<<dim3(4, 256), 256, 0, stream>>>(
        hh, fc2wp + (size_t)i * 1024 * 256, fc2_b + i * 256, y, y,
        T_TOK, 256, 1024);
  }

  out_kernel<<<32768, 256, 0, stream>>>(y, (float*)d_out);
}
